// NodeModel_86045374808683
// MI455X (gfx1250) — compile-verified, matrix operand order checked by reading, not run on hardware
//
#include <hip/hip_runtime.h>

// ---------------------------------------------------------------------------
// CDNA5 (gfx1250) WMMA GNN layer.
//   y   = relu(x @ W1a^T) @ W1b^T          (node MLP; commutes with gather)
//   s   = segment_sum(y[row], col); agg = LN1(s / max(cnt,1))
//   fx  = LN2(x + (x - agg) * w)
//   out = relu([fx, agg] @ W2a^T) @ W2b^T
// Weights are pre-packed once per launch into WMMA B-operand lane order
// (bf16), so the hot GEMM waves do zero weight conversion work.
// ---------------------------------------------------------------------------

typedef __attribute__((ext_vector_type(16))) __bf16 bf16x16;
typedef __attribute__((ext_vector_type(8)))  float  f32x8;

#define DEV __device__ __forceinline__

// WMMA 16-bit A/B operand layout (wave32), one 16x32 (rows x K) tile:
//   lane<16 : row = lane,    K = k0+[0..7]  and k0+[16..23]
//   lane>=16: row = lane-16, K = k0+[8..15] and k0+[24..31]
// B operand for X @ W^T is W itself in this layout (B[k][n] = W[n][k]).

DEV bf16x16 load_ab_f32(const float* mat, int ld, int row_base, int nrows,
                        int k0, int lane) {
  int m = row_base + (lane & 15);
  m = (m < nrows) ? m : (nrows - 1);           // clamp for tail tiles
  int kb = (lane >> 4) << 3;                   // 0 or 8
  const float* p = mat + (size_t)m * ld + k0 + kb;
  bf16x16 r;
#pragma unroll
  for (int i = 0; i < 8; ++i) r[i] = (__bf16)p[i];
#pragma unroll
  for (int i = 0; i < 8; ++i) r[i + 8] = (__bf16)p[16 + i];
  return r;
}

DEV bf16x16 load_ab_bf16(const __bf16* mat, int ld, int row_base, int nrows,
                         int k0, int lane) {
  int m = row_base + (lane & 15);
  m = (m < nrows) ? m : (nrows - 1);
  int kb = (lane >> 4) << 3;
  const __bf16* p = mat + (size_t)m * ld + k0 + kb;
  bf16x16 r;
#pragma unroll
  for (int i = 0; i < 8; ++i) r[i] = p[i];
#pragma unroll
  for (int i = 0; i < 8; ++i) r[i + 8] = p[16 + i];
  return r;
}

// Pre-packed weight chunk: one contiguous 32B bf16x16 per lane.
DEV bf16x16 load_w_packed(const __bf16* wp, int chunk, int lane) {
  return *reinterpret_cast<const bf16x16*>(wp +
         (((size_t)chunk * 32 + lane) << 4));
}

DEV f32x8 wmma_bf16(bf16x16 a, bf16x16 b, f32x8 c) {
  // (neg_a, A, neg_b, B, c_mod, C, reuse_a, reuse_b)
  return __builtin_amdgcn_wmma_f32_16x16x32_bf16(false, a, false, b,
                                                 (short)0, c, false, false);
}

// C/D tile layout: VGPR r holds rows r (lanes 0-15) / r+8 (lanes 16-31),
// column = lane & 15.
DEV void store_c_relu_bf16_lds(__bf16* mat, int ld, int lane, f32x8 acc) {
  int col = lane & 15;
  int rb  = (lane >> 4) << 3;
#pragma unroll
  for (int r = 0; r < 8; ++r) {
    float v = acc[r];
    v = v > 0.0f ? v : 0.0f;
    mat[(rb + r) * ld + col] = (__bf16)v;
  }
}

DEV void store_c_f32(float* mat, int ld, int row_base, int nrows, int lane,
                     f32x8 acc) {
  int col = lane & 15;
  int rb  = (lane >> 4) << 3;
#pragma unroll
  for (int r = 0; r < 8; ++r) {
    int m = row_base + rb + r;
    if (m < nrows) mat[(size_t)m * ld + col] = acc[r];
  }
}

// ------------------------------- kernels -----------------------------------

__global__ void __launch_bounds__(256) zero_kernel(float* p, long long count) {
  long long i = (long long)blockIdx.x * blockDim.x + threadIdx.x;
  long long stride = (long long)gridDim.x * blockDim.x;
  for (; i < count; i += stride) p[i] = 0.0f;
}

// Swizzle a [rows, K] f32 weight matrix into WMMA B-operand chunks:
// out[((nt*ksteps + ks)*32 + lane)*16 + e], chunk covers n-tile nt, k-step ks.
__global__ void __launch_bounds__(256) packw_kernel(
    const float* __restrict__ W, unsigned short* __restrict__ out_raw,
    int K, int nelem) {
  __bf16* out = reinterpret_cast<__bf16*>(out_raw);
  int i = blockIdx.x * blockDim.x + threadIdx.x;
  if (i >= nelem) return;
  int e     = i & 15;
  int lane  = (i >> 4) & 31;
  int chunk = i >> 9;                    // 512 elements per chunk
  int ksteps = K >> 5;
  int nt = chunk / ksteps;
  int ks = chunk % ksteps;
  int row = nt * 16 + (lane & 15);
  int kb  = (lane >> 4) << 3;
  int k   = ks * 32 + kb + ((e < 8) ? e : (e + 8));  // e>=8 -> kb+16+(e-8)
  out[i] = (__bf16)W[(size_t)row * K + k];
}

// y = relu(x @ W1a^T) @ W1b^T, one 16-row tile per wave. D = 64.
__global__ void __launch_bounds__(128) mlp1_kernel(
    const float* __restrict__ x, const unsigned short* __restrict__ w1a_raw,
    const unsigned short* __restrict__ w1b_raw, float* __restrict__ y,
    int n, int ntiles) {
  const __bf16* w1a = reinterpret_cast<const __bf16*>(w1a_raw);
  const __bf16* w1b = reinterpret_cast<const __bf16*>(w1b_raw);
  __shared__ __bf16 stage[4][16 * 64];
  int wave = threadIdx.x >> 5;
  int lane = threadIdx.x & 31;
  int tile = blockIdx.x * 4 + wave;
  if (tile >= ntiles) return;                 // whole-wave exit: EXEC all-1
  int row0 = tile * 16;

  bf16x16 a0 = load_ab_f32(x, 64, row0, n, 0,  lane);
  bf16x16 a1 = load_ab_f32(x, 64, row0, n, 32, lane);

  __bf16* st = stage[wave];
#pragma unroll
  for (int nt = 0; nt < 4; ++nt) {
    bf16x16 b0 = load_w_packed(w1a, nt * 2 + 0, lane);
    bf16x16 b1 = load_w_packed(w1a, nt * 2 + 1, lane);
    f32x8 acc = {};
    acc = wmma_bf16(a0, b0, acc);
    acc = wmma_bf16(a1, b1, acc);
    store_c_relu_bf16_lds(st + nt * 16, 64, lane, acc);
  }
  // second GEMM: A comes from this wave's private LDS slice (same-wave DS
  // ordering, no barrier needed)
  bf16x16 c0 = load_ab_bf16(st, 64, 0, 16, 0,  lane);
  bf16x16 c1 = load_ab_bf16(st, 64, 0, 16, 32, lane);
#pragma unroll
  for (int nt = 0; nt < 4; ++nt) {
    bf16x16 b0 = load_w_packed(w1b, nt * 2 + 0, lane);
    bf16x16 b1 = load_w_packed(w1b, nt * 2 + 1, lane);
    f32x8 acc = {};
    acc = wmma_bf16(c0, b0, acc);
    acc = wmma_bf16(c1, b1, acc);
    store_c_f32(y + nt * 16, 64, row0, n, lane, acc);
  }
}

// scatter-add: one wave per edge, 2 floats per lane. cnt bumped by lane 0.
__global__ void __launch_bounds__(256) scatter_kernel(
    const float* __restrict__ y, const int* __restrict__ row,
    const int* __restrict__ col, float* __restrict__ s,
    float* __restrict__ cnt, int E) {
  long long tid = (long long)blockIdx.x * blockDim.x + threadIdx.x;
  int e = (int)(tid >> 5);
  int lane = (int)(tid & 31);
  if (e >= E) return;
  int r = row[e];
  int c = col[e];
  const float2 v = reinterpret_cast<const float2*>(y + (size_t)r * 64)[lane];
  float* sd = s + (size_t)c * 64 + lane * 2;
  atomicAdd(sd,     v.x);
  atomicAdd(sd + 1, v.y);
  if (lane == 0) atomicAdd(cnt + c, 1.0f);
}

// per-node: agg = LN1(s/max(cnt,1)); fx = LN2(x + (x-agg)*w); h = [fx, agg]
__global__ void __launch_bounds__(256) ln_kernel(
    const float* __restrict__ x, const float* __restrict__ s,
    const float* __restrict__ cnt, const float* __restrict__ gamma1,
    const float* __restrict__ beta1, const float* __restrict__ w,
    const float* __restrict__ gamma2, const float* __restrict__ beta2,
    unsigned short* __restrict__ h_raw, int n) {
  __bf16* h = reinterpret_cast<__bf16*>(h_raw);
  int wave = threadIdx.x >> 5;
  int lane = threadIdx.x & 31;
  int node = blockIdx.x * 8 + wave;
  if (node >= n) return;
  int d0 = lane, d1 = lane + 32;
  const size_t base = (size_t)node * 64;

  float inv = 1.0f / fmaxf(cnt[node], 1.0f);
  float a0 = s[base + d0] * inv;
  float a1 = s[base + d1] * inv;

  // LayerNorm 1 (wave32 reduction over 64 features)
  float sum = a0 + a1;
#pragma unroll
  for (int m = 16; m >= 1; m >>= 1) sum += __shfl_xor(sum, m, 32);
  float mean = sum * (1.0f / 64.0f);
  float e0 = a0 - mean, e1 = a1 - mean;
  float vs = e0 * e0 + e1 * e1;
#pragma unroll
  for (int m = 16; m >= 1; m >>= 1) vs += __shfl_xor(vs, m, 32);
  float istd = rsqrtf(vs * (1.0f / 64.0f) + 1e-5f);
  float agg0 = e0 * istd * gamma1[d0] + beta1[d0];
  float agg1 = e1 * istd * gamma1[d1] + beta1[d1];

  float x0 = x[base + d0], x1 = x[base + d1];
  float f0 = x0 + (x0 - agg0) * w[d0];
  float f1 = x1 + (x1 - agg1) * w[d1];

  // LayerNorm 2
  float sum2 = f0 + f1;
#pragma unroll
  for (int m = 16; m >= 1; m >>= 1) sum2 += __shfl_xor(sum2, m, 32);
  float mean2 = sum2 * (1.0f / 64.0f);
  float g0 = f0 - mean2, g1 = f1 - mean2;
  float vs2 = g0 * g0 + g1 * g1;
#pragma unroll
  for (int m = 16; m >= 1; m >>= 1) vs2 += __shfl_xor(vs2, m, 32);
  float istd2 = rsqrtf(vs2 * (1.0f / 64.0f) + 1e-5f);
  float fx0 = g0 * istd2 * gamma2[d0] + beta2[d0];
  float fx1 = g1 * istd2 * gamma2[d1] + beta2[d1];

  __bf16* hr = h + (size_t)node * 128;
  hr[d0]      = (__bf16)fx0;
  hr[d1]      = (__bf16)fx1;
  hr[64 + d0] = (__bf16)agg0;
  hr[64 + d1] = (__bf16)agg1;
}

// out = relu(h @ W2a^T) @ W2b^T ; h: [N,128] bf16, W2a: [64,128], W2b: [64,64]
__global__ void __launch_bounds__(128) mlp2_kernel(
    const unsigned short* __restrict__ h_raw,
    const unsigned short* __restrict__ w2a_raw,
    const unsigned short* __restrict__ w2b_raw,
    float* __restrict__ out, int n, int ntiles) {
  const __bf16* h   = reinterpret_cast<const __bf16*>(h_raw);
  const __bf16* w2a = reinterpret_cast<const __bf16*>(w2a_raw);
  const __bf16* w2b = reinterpret_cast<const __bf16*>(w2b_raw);
  __shared__ __bf16 stage[4][16 * 64];
  int wave = threadIdx.x >> 5;
  int lane = threadIdx.x & 31;
  int tile = blockIdx.x * 4 + wave;
  if (tile >= ntiles) return;
  int row0 = tile * 16;

  bf16x16 a[4];
#pragma unroll
  for (int ks = 0; ks < 4; ++ks)
    a[ks] = load_ab_bf16(h, 128, row0, n, ks * 32, lane);

  __bf16* st = stage[wave];
#pragma unroll
  for (int nt = 0; nt < 4; ++nt) {
    f32x8 acc = {};
#pragma unroll
    for (int ks = 0; ks < 4; ++ks) {
      bf16x16 b = load_w_packed(w2a, nt * 4 + ks, lane);
      acc = wmma_bf16(a[ks], b, acc);
    }
    store_c_relu_bf16_lds(st + nt * 16, 64, lane, acc);
  }

  bf16x16 c0 = load_ab_bf16(st, 64, 0, 16, 0,  lane);
  bf16x16 c1 = load_ab_bf16(st, 64, 0, 16, 32, lane);
#pragma unroll
  for (int nt = 0; nt < 4; ++nt) {
    bf16x16 b0 = load_w_packed(w2b, nt * 2 + 0, lane);
    bf16x16 b1 = load_w_packed(w2b, nt * 2 + 1, lane);
    f32x8 acc = {};
    acc = wmma_bf16(c0, b0, acc);
    acc = wmma_bf16(c1, b1, acc);
    store_c_f32(out + nt * 16, 64, row0, n, lane, acc);
  }
}

// ------------------------------ launcher -----------------------------------

extern "C" void kernel_launch(void* const* d_in, const int* in_sizes, int n_in,
                              void* d_out, int out_size, void* d_ws,
                              size_t ws_size, hipStream_t stream) {
  const float* x      = (const float*)d_in[0];
  const int*   ei     = (const int*)d_in[1];   // [2, E] flat: row then col
  const float* W1a    = (const float*)d_in[2];
  const float* W1b    = (const float*)d_in[3];
  const float* gamma1 = (const float*)d_in[4];
  const float* beta1  = (const float*)d_in[5];
  const float* w      = (const float*)d_in[6];
  const float* gamma2 = (const float*)d_in[7];
  const float* beta2  = (const float*)d_in[8];
  const float* W2a    = (const float*)d_in[9];
  const float* W2b    = (const float*)d_in[10];

  const int N = in_sizes[0] / 64;
  const int E = in_sizes[1] / 2;
  const int* row = ei;
  const int* col = ei + E;

  // workspace layout (all offsets 32B aligned for N multiple of 8)
  float* y   = (float*)d_ws;                        // [N,64]  f32
  float* s   = y + (size_t)N * 64;                  // [N,64]  f32
  float* cnt = s + (size_t)N * 64;                  // [N]     f32
  unsigned short* h    = (unsigned short*)(cnt + N);// [N,128] bf16
  unsigned short* w1a_p = h + (size_t)N * 128;      // 4096 bf16
  unsigned short* w1b_p = w1a_p + 4096;             // 4096 bf16
  unsigned short* w2a_p = w1b_p + 4096;             // 8192 bf16
  unsigned short* w2b_p = w2a_p + 8192;             // 4096 bf16

  float* out = (float*)d_out;
  const int ntiles = (N + 15) / 16;

  // 1) zero s and cnt (contiguous)
  long long zcount = (long long)N * 65;
  zero_kernel<<<2048, 256, 0, stream>>>(s, zcount);

  // 1b) pack weights into WMMA B-operand lane order (bf16)
  packw_kernel<<<16, 256, 0, stream>>>(W1a, w1a_p, 64, 4096);
  packw_kernel<<<16, 256, 0, stream>>>(W1b, w1b_p, 64, 4096);
  packw_kernel<<<32, 256, 0, stream>>>(W2a, w2a_p, 128, 8192);
  packw_kernel<<<16, 256, 0, stream>>>(W2b, w2b_p, 64, 4096);

  // 2) node MLP1 -> y
  mlp1_kernel<<<(ntiles + 3) / 4, 128, 0, stream>>>(x, w1a_p, w1b_p, y, N,
                                                    ntiles);

  // 3) scatter-add y[row] into s[col], count edges per node
  long long sthreads = (long long)E * 32;
  scatter_kernel<<<(unsigned)((sthreads + 255) / 256), 256, 0, stream>>>(
      y, row, col, s, cnt, E);

  // 4) layernorms + repulsion -> h = [fx, agg] (bf16)
  ln_kernel<<<(N + 7) / 8, 256, 0, stream>>>(x, s, cnt, gamma1, beta1, w,
                                             gamma2, beta2, h, N);

  // 5) node MLP2 -> out
  mlp2_kernel<<<(ntiles + 3) / 4, 128, 0, stream>>>(h, w2a_p, w2b_p, out, N,
                                                    ntiles);
}
